// TimePoseFunction_2774548873477
// MI455X (gfx1250) — matrix operand order repaired and checked by reading, not compiled, not run on hardware
//
#include <hip/hip_runtime.h>
#include <hip/hip_bf16.h>

// ---------------------------------------------------------------------------
// TimePose MLP for MI455X (gfx1250, wave32, WMMA).
//   interp -> [gemm+BN-stats -> bn-finalize] x6 -> head
// GEMMs use v_wmma_f32_16x16x32_f16 (f16 A/B staged in LDS, f32 accum).
// BatchNorm batch statistics fused into GEMM epilogue (LDS ds_add_f32 +
// per-channel global atomics); normalization + LeakyReLU fused into the
// next layer's A-tile load (single FMA with precomputed scale/shift).
// ---------------------------------------------------------------------------

#define N_SAMP      524288
#define NCH         128
#define BASE_STRIDE 44      // [t, feat0(20), feat1(20), pad(3)]
#define MBLK        128     // rows per block
#define AS_STRIDE   72      // (64 panel cols + 8 pad) halves, keeps 16B align

typedef __attribute__((ext_vector_type(16))) _Float16 v16h;
typedef __attribute__((ext_vector_type(8)))  float    v8f;

__device__ __forceinline__ v16h ld_frag(const _Float16* p0, const _Float16* p1) {
    // Two contiguous 16B LDS chunks -> one WMMA operand (ds_load_b128 x2).
    v16h r;
    ((float4*)&r)[0] = *(const float4*)p0;
    ((float4*)&r)[1] = *(const float4*)p1;
    return r;
}

// ---------------------------------------------------------------------------
// Hash-grid interpolation: base[s] = [t, f0(20), f1(20), 0,0,0]
// idx = (t_mid ^ K) mod 2^19  (Python mod == bitwise AND for pow2 modulus)
// ---------------------------------------------------------------------------
__global__ __launch_bounds__(256) void interp_kernel(
    const float* __restrict__ t_in, const float* __restrict__ fd0,
    const float* __restrict__ fd1, float* __restrict__ base) {
    const int s = blockIdx.x * 256 + threadIdx.x;
    const float tv = t_in[s];
    const float tf = floorf(tv);
    const long long tm = (long long)tf;
    const float fr = tv - tf;
    const float lp = fr * (fr - 1.0f) * 0.5f;
    const float lm = -(fr + 1.0f) * (fr - 1.0f);
    const float ln = (fr + 1.0f) * fr * 0.5f;
    const long long H0 = 2654435762ll;   // 2654435761 + 1 (xor binds loose)
    const unsigned msk = 0x7FFFFu;       // mod 524288
    const unsigned ip0 = ((unsigned)((tm - 1) ^ H0)) & msk;
    const unsigned im0 = ((unsigned)((tm    ) ^ H0)) & msk;
    const unsigned in0 = ((unsigned)((tm + 1) ^ H0)) & msk;
    const unsigned ip1 = ((unsigned)((tm - 1) ^ 1ll)) & msk;
    const unsigned im1 = ((unsigned)((tm    ) ^ 1ll)) & msk;
    const unsigned in1 = ((unsigned)((tm + 1) ^ 1ll)) & msk;

    float* bp = base + (size_t)s * BASE_STRIDE;
    bp[0] = tv;
    // rows are 20 floats = 80B -> 16B aligned: 5x float4 gathers per row
    const float4* a0 = (const float4*)(fd0 + (size_t)ip0 * 20);
    const float4* b0 = (const float4*)(fd0 + (size_t)im0 * 20);
    const float4* c0 = (const float4*)(fd0 + (size_t)in0 * 20);
    const float4* a1 = (const float4*)(fd1 + (size_t)ip1 * 20);
    const float4* b1 = (const float4*)(fd1 + (size_t)im1 * 20);
    const float4* c1 = (const float4*)(fd1 + (size_t)in1 * 20);
#pragma unroll
    for (int j = 0; j < 5; ++j) {
        float4 p = a0[j], m = b0[j], n = c0[j];
        bp[1 + 4 * j + 0] = lp * p.x + lm * m.x + ln * n.x;
        bp[1 + 4 * j + 1] = lp * p.y + lm * m.y + ln * n.y;
        bp[1 + 4 * j + 2] = lp * p.z + lm * m.z + ln * n.z;
        bp[1 + 4 * j + 3] = lp * p.w + lm * m.w + ln * n.w;
    }
#pragma unroll
    for (int j = 0; j < 5; ++j) {
        float4 p = a1[j], m = b1[j], n = c1[j];
        bp[21 + 4 * j + 0] = lp * p.x + lm * m.x + ln * n.x;
        bp[21 + 4 * j + 1] = lp * p.y + lm * m.y + ln * n.y;
        bp[21 + 4 * j + 2] = lp * p.z + lm * m.z + ln * n.z;
        bp[21 + 4 * j + 3] = lp * p.w + lm * m.w + ln * n.w;
    }
    bp[41] = 0.f; bp[42] = 0.f; bp[43] = 0.f;
}

// ---------------------------------------------------------------------------
// GEMM layer: h = x @ W + b  (x built on the fly: BN+LeakyReLU of prev layer
// and/or base features), h -> f16, per-channel sum/sum^2 -> global atomics.
// MODE 0: x = act(hin)            DIN=128 KP=128
// MODE 1: x = base[0:41]          DIN=41  KP=64
// MODE 2: x = [base[0:41], act(hin)] DIN=169 KP=192
// Block: 256 thr = 8 waves; each wave owns 16 rows x 128 cols (8 acc tiles).
// ---------------------------------------------------------------------------
template <int MODE, int DIN, int KP>
__global__ __launch_bounds__(256) void gemm_bn_kernel(
    const float* __restrict__ base, const _Float16* __restrict__ hin,
    const float* __restrict__ scale, const float* __restrict__ shift,
    const float* __restrict__ W, const float* __restrict__ bias,
    _Float16* __restrict__ hout, float* __restrict__ ssum,
    float* __restrict__ ssq) {
    extern __shared__ char smem_raw[];
    _Float16* As = (_Float16*)smem_raw;          // [128][AS_STRIDE] activations (panel)
    _Float16* Bs = As + MBLK * AS_STRIDE;        // [128][AS_STRIDE] weights, transposed
    float* rsum = (float*)(Bs + NCH * AS_STRIDE);
    float* rsq  = rsum + NCH;
    float* ssc  = rsq + NCH;
    float* ssh  = ssc + NCH;

    const int tid  = threadIdx.x;
    const int lane = tid & 31;
    const int wv   = tid >> 5;
    const int l15  = lane & 15;
    const int hi   = lane >> 4;
    const size_t row0 = (size_t)blockIdx.x * MBLK;

    if (tid < NCH) {
        rsum[tid] = 0.f; rsq[tid] = 0.f;
        if (MODE != 1) { ssc[tid] = scale[tid]; ssh[tid] = shift[tid]; }
    }
    __syncthreads();

    v8f acc[8] = {};
    // A-fragment per §7.12.2: lanes 0-15 hold K[0:8)+K[16:24), 16-31 the other
    // 8-half chunks; row stride 72 halves keeps every chunk 16B aligned.
    const _Float16* aRow = As + (wv * 16 + l15) * AS_STRIDE + hi * 8;

    for (int kp = 0; kp < KP; kp += 64) {
        // ---- stage A panel (128 rows x 64 K) as f16, fusing BN + LeakyReLU
        for (int e = tid; e < MBLK * 64; e += 256) {
            const int r = e >> 6, c = e & 63, kk = kp + c;
            float v;
            if (MODE == 0) {
                float hv = (float)hin[(row0 + r) * NCH + kk];
                v = ssc[kk] * hv + ssh[kk];
                v = v > 0.f ? v : 0.01f * v;
            } else if (MODE == 1) {
                v = (kk < DIN) ? base[(row0 + r) * BASE_STRIDE + kk] : 0.f;
            } else {
                if (kk < 41) {
                    v = base[(row0 + r) * BASE_STRIDE + kk];
                } else if (kk < DIN) {
                    const int c2 = kk - 41;
                    float hv = (float)hin[(row0 + r) * NCH + c2];
                    v = ssc[c2] * hv + ssh[c2];
                    v = v > 0.f ? v : 0.01f * v;
                } else {
                    v = 0.f;
                }
            }
            As[r * AS_STRIDE + c] = (_Float16)v;
        }
        // ---- stage W panel transposed: Bs[col][k] so each of a lane's two 8-K chunks is
        // contiguous (two ds_load_b128 per fragment)
        for (int e = tid; e < 64 * NCH; e += 256) {
            const int k = e >> 7, c = e & 127, kk = kp + k;
            const float w = (kk < DIN) ? W[(size_t)kk * NCH + c] : 0.f;
            Bs[c * AS_STRIDE + k] = (_Float16)w;
        }
        __syncthreads();
#pragma unroll
        for (int ks = 0; ks < 64; ks += 32) {
            const v16h a = ld_frag(aRow + ks, aRow + ks + 16);
#pragma unroll
            for (int n = 0; n < 8; ++n) {
                const _Float16* bp = Bs + (n * 16 + l15) * AS_STRIDE + ks + hi * 8;
                const v16h b = ld_frag(bp, bp + 16);
                acc[n] = __builtin_amdgcn_wmma_f32_16x16x32_f16(
                    false, a, false, b, (short)0, acc[n], false, false);
            }
        }
        __syncthreads();
    }

    // ---- epilogue: bias add, f16 store, fused BN batch statistics
    // C/D layout: lane l, vgpr v -> row = v + 8*(l>=16), col = 16n + (l&15)
    const size_t rowa = row0 + wv * 16 + hi * 8;
#pragma unroll
    for (int n = 0; n < 8; ++n) {
        const int col = n * 16 + l15;
        const float bb = bias[col];
        float ps = 0.f, pq = 0.f;
#pragma unroll
        for (int v = 0; v < 8; ++v) {
            const float h = acc[n][v] + bb;
            hout[(rowa + v) * NCH + col] = (_Float16)h;
            ps += h; pq += h * h;
        }
        atomicAdd(&rsum[col], ps);   // ds_add_f32
        atomicAdd(&rsq[col], pq);
    }
    __syncthreads();
    if (tid < NCH) {
        atomicAdd(&ssum[tid], rsum[tid]);   // global_atomic_add_f32
        atomicAdd(&ssq[tid], rsq[tid]);
    }
}

// sums -> scale/shift (affine fold of BN), then re-zero stats for next layer
__global__ void bn_finalize_kernel(float* ssum, float* ssq,
                                   const float* __restrict__ gamma,
                                   const float* __restrict__ beta,
                                   float* scale, float* shift) {
    const int c = threadIdx.x;
    const float invN = 1.0f / (float)N_SAMP;
    const float mu  = ssum[c] * invN;
    const float var = ssq[c] * invN - mu * mu;   // biased variance
    const float inv = rsqrtf(var + 1e-5f);
    const float sc  = gamma[c] * inv;
    scale[c] = sc;
    shift[c] = beta[c] - mu * sc;
    ssum[c] = 0.f; ssq[c] = 0.f;
}

__global__ void zero_kernel(float* p) { p[threadIdx.x] = 0.f; }

// ---------------------------------------------------------------------------
// Head: act(h5) @ W_rot (normalize) and @ W_trans
// ---------------------------------------------------------------------------
__global__ __launch_bounds__(256) void head_kernel(
    const _Float16* __restrict__ h5, const float* __restrict__ scale,
    const float* __restrict__ shift, const float* __restrict__ Wr,
    const float* __restrict__ br, const float* __restrict__ Wt,
    const float* __restrict__ bt, float* __restrict__ outT,
    float* __restrict__ outR) {
    __shared__ float swr[NCH * 4];
    __shared__ float swt[NCH * 3];
    __shared__ float ssc[NCH], ssh[NCH];
    const int tid = threadIdx.x;
    if (tid < NCH) { ssc[tid] = scale[tid]; ssh[tid] = shift[tid]; }
    for (int e = tid; e < NCH * 4; e += 256) swr[e] = Wr[e];
    for (int e = tid; e < NCH * 3; e += 256) swt[e] = Wt[e];
    __syncthreads();

    const size_t s = (size_t)blockIdx.x * 256 + tid;
    const _Float16* hp = h5 + s * NCH;
    float r0 = 0, r1 = 0, r2 = 0, r3 = 0, t0 = 0, t1 = 0, t2 = 0;
#pragma unroll 4
    for (int c = 0; c < NCH; ++c) {
        float a = ssc[c] * (float)hp[c] + ssh[c];
        a = a > 0.f ? a : 0.01f * a;
        r0 += a * swr[c * 4 + 0]; r1 += a * swr[c * 4 + 1];
        r2 += a * swr[c * 4 + 2]; r3 += a * swr[c * 4 + 3];
        t0 += a * swt[c * 3 + 0]; t1 += a * swt[c * 3 + 1];
        t2 += a * swt[c * 3 + 2];
    }
    r0 += br[0]; r1 += br[1]; r2 += br[2]; r3 += br[3];
    t0 += bt[0]; t1 += bt[1]; t2 += bt[2];
    const float inv = rsqrtf(r0 * r0 + r1 * r1 + r2 * r2 + r3 * r3);
    outT[s * 3 + 0] = t0; outT[s * 3 + 1] = t1; outT[s * 3 + 2] = t2;
    outR[s * 4 + 0] = r0 * inv; outR[s * 4 + 1] = r1 * inv;
    outR[s * 4 + 2] = r2 * inv; outR[s * 4 + 3] = r3 * inv;
}

// ---------------------------------------------------------------------------
extern "C" void kernel_launch(void* const* d_in, const int* in_sizes, int n_in,
                              void* d_out, int out_size, void* d_ws,
                              size_t ws_size, hipStream_t stream) {
    (void)in_sizes; (void)n_in; (void)out_size; (void)ws_size;
    // setup_inputs order: t, fd0, fd1, Ws[0..5], bs[0..5], gammas[0..5],
    // betas[0..5], W_rot, b_rot, W_trans, b_trans
    const float* t   = (const float*)d_in[0];
    const float* fd0 = (const float*)d_in[1];
    const float* fd1 = (const float*)d_in[2];
    const float* Ws[6]; const float* bs[6]; const float* gs[6]; const float* be[6];
    for (int i = 0; i < 6; ++i) {
        Ws[i] = (const float*)d_in[3 + i];
        bs[i] = (const float*)d_in[9 + i];
        gs[i] = (const float*)d_in[15 + i];
        be[i] = (const float*)d_in[21 + i];
    }
    const float* Wr = (const float*)d_in[27];
    const float* br = (const float*)d_in[28];
    const float* Wt = (const float*)d_in[29];
    const float* bt = (const float*)d_in[30];

    // workspace: base f32 [N,44] | h0 f16 [N,128] | h1 f16 [N,128] | stats(512 f32)
    char* w = (char*)d_ws;
    float* base = (float*)w;      w += (size_t)N_SAMP * BASE_STRIDE * sizeof(float);
    _Float16* h0 = (_Float16*)w;  w += (size_t)N_SAMP * NCH * sizeof(_Float16);
    _Float16* h1 = (_Float16*)w;  w += (size_t)N_SAMP * NCH * sizeof(_Float16);
    float* ssum  = (float*)w;
    float* ssq   = ssum + NCH;
    float* scale = ssum + 2 * NCH;
    float* shift = ssum + 3 * NCH;

    float* outT = (float*)d_out;                 // trans [N,3]
    float* outR = outT + (size_t)N_SAMP * 3;     // rot   [N,4]

    const dim3 blk(256);
    const dim3 grid(N_SAMP / MBLK);
    const size_t lds = (size_t)(MBLK + NCH) * AS_STRIDE * sizeof(_Float16) +
                       4 * NCH * sizeof(float);

    zero_kernel<<<1, 256, 0, stream>>>(ssum);  // zeros ssum+ssq (contiguous)
    interp_kernel<<<N_SAMP / 256, blk, 0, stream>>>(t, fd0, fd1, base);

    gemm_bn_kernel<1, 41, 64><<<grid, blk, lds, stream>>>(
        base, nullptr, nullptr, nullptr, Ws[0], bs[0], h0, ssum, ssq);
    bn_finalize_kernel<<<1, NCH, 0, stream>>>(ssum, ssq, gs[0], be[0], scale, shift);

    gemm_bn_kernel<0, 128, 128><<<grid, blk, lds, stream>>>(
        nullptr, h0, scale, shift, Ws[1], bs[1], h1, ssum, ssq);
    bn_finalize_kernel<<<1, NCH, 0, stream>>>(ssum, ssq, gs[1], be[1], scale, shift);

    gemm_bn_kernel<0, 128, 128><<<grid, blk, lds, stream>>>(
        nullptr, h1, scale, shift, Ws[2], bs[2], h0, ssum, ssq);
    bn_finalize_kernel<<<1, NCH, 0, stream>>>(ssum, ssq, gs[2], be[2], scale, shift);

    gemm_bn_kernel<2, 169, 192><<<grid, blk, lds, stream>>>(
        base, h0, scale, shift, Ws[3], bs[3], h1, ssum, ssq);
    bn_finalize_kernel<<<1, NCH, 0, stream>>>(ssum, ssq, gs[3], be[3], scale, shift);

    gemm_bn_kernel<0, 128, 128><<<grid, blk, lds, stream>>>(
        nullptr, h1, scale, shift, Ws[4], bs[4], h0, ssum, ssq);
    bn_finalize_kernel<<<1, NCH, 0, stream>>>(ssum, ssq, gs[4], be[4], scale, shift);

    gemm_bn_kernel<0, 128, 128><<<grid, blk, lds, stream>>>(
        nullptr, h0, scale, shift, Ws[5], bs[5], h1, ssum, ssq);
    bn_finalize_kernel<<<1, NCH, 0, stream>>>(ssum, ssq, gs[5], be[5], scale, shift);

    head_kernel<<<N_SAMP / 256, blk, 0, stream>>>(h1, scale, shift, Wr, br, Wt, bt,
                                                  outT, outR);
}
